// AdaptiveConditionedGraphTransformer_83339545412145
// MI455X (gfx1250) — matrix operand order repaired and checked by reading, not compiled, not run on hardware
//
#include <hip/hip_runtime.h>
#include <hip/hip_bf16.h>

// ---------------- problem constants (match reference) ----------------
#define N_NODES 1024
#define N_EDGES 16384
#define FEAT    10
#define DMODEL  256
#define NHEAD   4
#define DHEADC  256          // tconv per-head dim D
#define HD      1024         // NHEAD * DHEADC
#define NGEN    20
#define KKNOWN  10
#define FFDIM   2048
#define DH      64           // encoder head dim

typedef __attribute__((ext_vector_type(16))) __bf16 v16bf;
typedef __attribute__((ext_vector_type(8)))  float  v8f;
typedef __attribute__((ext_vector_type(4)))  unsigned u32x4;

union FragBF { v16bf v; u32x4 q[2]; };

__device__ __forceinline__ unsigned short f2bf(float f) {
  unsigned u = __float_as_uint(f);
  u += 0x7FFFu + ((u >> 16) & 1u);      // round-to-nearest-even
  return (unsigned short)(u >> 16);
}
__device__ __forceinline__ unsigned pack2bf(float lo, float hi) {
  return ((unsigned)f2bf(hi) << 16) | (unsigned)f2bf(lo);
}
__device__ __forceinline__ unsigned fkey(float f) {   // order-preserving float->uint
  unsigned u = __float_as_uint(f);
  return (u & 0x80000000u) ? ~u : (u | 0x80000000u);
}
__device__ __forceinline__ float funkey(unsigned k) {
  unsigned u = (k & 0x80000000u) ? (k ^ 0x80000000u) : ~k;
  return __uint_as_float(u);
}

// ---------------- weight convert+transpose: W[K][N] f32 -> Bt[Npad][Kpad] bf16 ----
__global__ void wconvert_kernel(const float* __restrict__ W, int ldw, int rows, int cols,
                                unsigned short* __restrict__ Bt, int Kpad, int Npad)
{
  int i = blockIdx.x * blockDim.x + threadIdx.x;
  if (i >= Npad * Kpad) return;
  int n = i / Kpad, k = i - n * Kpad;
  float v = (n < cols && k < rows) ? W[(long)k * ldw + n] : 0.0f;
  Bt[(long)n * Kpad + k] = f2bf(v);
}

// ---------------- FAST bf16 WMMA GEMM (full tiles only) ----------------
// Requires: M%64==0 (grid.y), N%128==0 (grid.x), Kdim%32==0.
// A: f32 activations (row major, lda, 16B-aligned rows in fast calls).
// Bt: pre-converted bf16 weight, [Npad][Kpad] row major (n-major = transposed).
// Block tile 64x128, 8 waves, each wave 32x32 as 2x2 v_wmma_f32_16x16x32_bf16.
// Epilogue behavior fixed at compile time: +residual (ADD) / ReLU.
template <bool HAS_ADD, bool RELU>
__global__ __launch_bounds__(256) void gemm_bf16_fast(
    const float* __restrict__ A, int lda,
    const unsigned short* __restrict__ Bt, int Kpad,
    const float* __restrict__ bias,
    const float* __restrict__ add, int ldadd,
    float* __restrict__ C, int ldc,
    int Kdim)
{
  __shared__ unsigned short lA[64 * 32];   // [m][k], 64B rows
  __shared__ unsigned short lB[128 * 40];  // [n][k], 80B rows (16B aligned)
  unsigned* lA32 = reinterpret_cast<unsigned*>(lA);
  unsigned* lB32 = reinterpret_cast<unsigned*>(lB);

  const int tid  = threadIdx.x;
  const int lane = tid & 31;
  const int wv   = tid >> 5;           // 0..7
  const int wm   = (wv >> 2) * 32;     // 0 / 32
  const int wn   = (wv & 3) * 32;      // 0..96
  const int mB   = blockIdx.y * 64;
  const int nB   = blockIdx.x * 128;

  v8f acc[2][2];
  #pragma unroll
  for (int i = 0; i < 2; ++i)
    #pragma unroll
    for (int j = 0; j < 2; ++j)
      #pragma unroll
      for (int r = 0; r < 8; ++r) acc[i][j][r] = 0.0f;

  const int ar = tid >> 2;             // A: row 0..63
  const int ak = (tid & 3) * 8;        // A: k chunk of 8
  const int bn = tid >> 1;             // B: row (n) 0..127
  const int bh = (tid & 1) * 2;        // B: which 16B pair of the 64B row

  for (int k0 = 0; k0 < Kdim; k0 += 32) {
    // --- stage A 64x32: 2x float4 load, convert, 4x packed b32 store ---
    {
      const float4* ap = reinterpret_cast<const float4*>(A + (long)(mB + ar) * lda + k0 + ak);
      float4 x0 = ap[0];
      float4 x1 = ap[1];
      unsigned* dp = &lA32[ar * 16 + (ak >> 1)];
      dp[0] = pack2bf(x0.x, x0.y);
      dp[1] = pack2bf(x0.z, x0.w);
      dp[2] = pack2bf(x1.x, x1.y);
      dp[3] = pack2bf(x1.z, x1.w);
    }
    // --- stage B 128x32 bf16: pure copy, 2x b128 load + 2x b128 LDS store ---
    {
      const u32x4* src = reinterpret_cast<const u32x4*>(Bt + (long)(nB + bn) * Kpad + k0);
      u32x4 x0 = src[bh];
      u32x4 x1 = src[bh + 1];
      u32x4* dst = reinterpret_cast<u32x4*>(lB32 + bn * 20);
      dst[bh]     = x0;
      dst[bh + 1] = x1;
    }
    __syncthreads();

    // --- fragments per documented wave32 layouts ---
    FragBF fa[2], fb[2];
    {
      int mrow = lane & 15;
      int kbA  = (lane < 16) ? 0 : 8;    // A lane<16: K 0-7,16-23 ; else 8-15,24-31
      #pragma unroll
      for (int i = 0; i < 2; ++i) {
        const unsigned short* p = &lA[(wm + i * 16 + mrow) * 32];
        fa[i].q[0] = *reinterpret_cast<const u32x4*>(p + kbA);
        fa[i].q[1] = *reinterpret_cast<const u32x4*>(p + kbA + 16);
      }
      int kbB = (lane < 16) ? 0 : 8;     // B lane<16: K 0-7,16-23 ; else 8-15,24-31
      #pragma unroll
      for (int j = 0; j < 2; ++j) {
        const unsigned short* p = &lB[(wn + j * 16 + mrow) * 40 + kbB];
        fb[j].q[0] = *reinterpret_cast<const u32x4*>(p);
        fb[j].q[1] = *reinterpret_cast<const u32x4*>(p + 16);
      }
    }
    #pragma unroll
    for (int i = 0; i < 2; ++i)
      #pragma unroll
      for (int j = 0; j < 2; ++j)
        acc[i][j] = __builtin_amdgcn_wmma_f32_16x16x32_bf16(
            false, fa[i].v, false, fb[j].v, (short)0, acc[i][j], false, false);
    __syncthreads();
  }

  // --- epilogue (no bounds guards, no runtime flag checks) ---
  int colL = lane & 15;
  int row0 = (lane >> 4) * 8;
  #pragma unroll
  for (int i = 0; i < 2; ++i) {
    #pragma unroll
    for (int j = 0; j < 2; ++j) {
      int gcol = nB + wn + j * 16 + colL;
      float bv = bias[gcol];
      #pragma unroll
      for (int r = 0; r < 8; ++r) {
        int grow = mB + wm + i * 16 + row0 + r;
        float v = acc[i][j][r] + bv;
        if (HAS_ADD) v += add[(long)grow * ldadd + gcol];
        if (RELU)    v = v > 0.0f ? v : 0.0f;
        C[(long)grow * ldc + gcol] = v;
      }
    }
  }
}

// ---------------- GENERIC guarded GEMM (ragged shapes; <1% of FLOPs) ----------
__global__ __launch_bounds__(256) void gemm_bf16_generic(
    const float* __restrict__ A, int lda,
    const float* __restrict__ B, int ldb,
    const float* __restrict__ bias,
    float* __restrict__ C, int ldc,
    int M, int Ncols, int Kdim)
{
  __shared__ unsigned short lA[64 * 32];
  __shared__ unsigned short lB[128 * 40];
  unsigned* lA32 = reinterpret_cast<unsigned*>(lA);
  unsigned* lB32 = reinterpret_cast<unsigned*>(lB);

  const int tid  = threadIdx.x;
  const int lane = tid & 31;
  const int wv   = tid >> 5;
  const int wm   = (wv >> 2) * 32;
  const int wn   = (wv & 3) * 32;
  const int mB   = blockIdx.y * 64;
  const int nB   = blockIdx.x * 128;

  v8f acc[2][2];
  #pragma unroll
  for (int i = 0; i < 2; ++i)
    #pragma unroll
    for (int j = 0; j < 2; ++j)
      #pragma unroll
      for (int r = 0; r < 8; ++r) acc[i][j][r] = 0.0f;

  const int ar  = tid >> 2;
  const int ak  = (tid & 3) * 8;
  const int krp = tid >> 4;
  const int bn0 = (tid & 15) * 8;

  for (int k0 = 0; k0 < Kdim; k0 += 32) {
    {
      int gm = mB + ar;
      #pragma unroll
      for (int u = 0; u < 4; ++u) {
        int gk = k0 + ak + 2 * u;
        float lo = (gm < M && gk     < Kdim) ? A[(long)gm * lda + gk]     : 0.0f;
        float hi = (gm < M && gk + 1 < Kdim) ? A[(long)gm * lda + gk + 1] : 0.0f;
        lA32[ar * 16 + (ak >> 1) + u] = pack2bf(lo, hi);
      }
    }
    {
      int gk0 = k0 + 2 * krp;
      #pragma unroll
      for (int u = 0; u < 8; ++u) {
        int gn = nB + bn0 + u;
        float lo = (gk0     < Kdim && gn < Ncols) ? B[(long)gk0 * ldb + gn]       : 0.0f;
        float hi = (gk0 + 1 < Kdim && gn < Ncols) ? B[(long)(gk0 + 1) * ldb + gn] : 0.0f;
        lB32[(bn0 + u) * 20 + krp] = pack2bf(lo, hi);
      }
    }
    __syncthreads();

    FragBF fa[2], fb[2];
    {
      int mrow = lane & 15;
      int kbA  = (lane < 16) ? 0 : 8;
      #pragma unroll
      for (int i = 0; i < 2; ++i) {
        const unsigned short* p = &lA[(wm + i * 16 + mrow) * 32];
        fa[i].q[0] = *reinterpret_cast<const u32x4*>(p + kbA);
        fa[i].q[1] = *reinterpret_cast<const u32x4*>(p + kbA + 16);
      }
      int kbB = (lane < 16) ? 0 : 8;
      #pragma unroll
      for (int j = 0; j < 2; ++j) {
        const unsigned short* p = &lB[(wn + j * 16 + mrow) * 40 + kbB];
        fb[j].q[0] = *reinterpret_cast<const u32x4*>(p);
        fb[j].q[1] = *reinterpret_cast<const u32x4*>(p + 16);
      }
    }
    #pragma unroll
    for (int i = 0; i < 2; ++i)
      #pragma unroll
      for (int j = 0; j < 2; ++j)
        acc[i][j] = __builtin_amdgcn_wmma_f32_16x16x32_bf16(
            false, fa[i].v, false, fb[j].v, (short)0, acc[i][j], false, false);
    __syncthreads();
  }

  int colL = lane & 15;
  int row0 = (lane >> 4) * 8;
  #pragma unroll
  for (int i = 0; i < 2; ++i) {
    #pragma unroll
    for (int j = 0; j < 2; ++j) {
      int gcol = nB + wn + j * 16 + colL;
      if (gcol >= Ncols) continue;
      float bv = bias[gcol];
      #pragma unroll
      for (int r = 0; r < 8; ++r) {
        int grow = mB + wm + i * 16 + row0 + r;
        if (grow >= M) continue;
        C[(long)grow * ldc + gcol] = acc[i][j][r] + bv;
      }
    }
  }
}

// ---------------- graph attention (tconv) edge kernels ----------------
__global__ __launch_bounds__(256) void edge_dot_kernel(
    const int* __restrict__ ei, const float* __restrict__ q,
    const float* __restrict__ k, float* __restrict__ sc,
    unsigned* __restrict__ mmax)
{
  int e    = blockIdx.x * 8 + (threadIdx.x >> 5);
  int lane = threadIdx.x & 31;
  if (e >= N_EDGES) return;
  int s = ei[e], d = ei[N_EDGES + e];
  #pragma unroll
  for (int h = 0; h < NHEAD; ++h) {
    const float* qp = q + (long)d * HD + h * DHEADC;
    const float* kp = k + (long)s * HD + h * DHEADC;
    float p = 0.0f;
    for (int x = lane; x < DHEADC; x += 32) p += qp[x] * kp[x];
    #pragma unroll
    for (int m = 16; m > 0; m >>= 1) p += __shfl_xor(p, m, 32);
    if (lane == 0) {
      p *= 0.0625f;                       // 1/sqrt(256)
      sc[e * NHEAD + h] = p;
      atomicMax(&mmax[d * NHEAD + h], fkey(p));   // segment max
    }
  }
}

__global__ void edge_exp_kernel(const int* __restrict__ ei,
                                const float* __restrict__ sc,
                                const unsigned* __restrict__ mmax,
                                float* __restrict__ aex,
                                float* __restrict__ zsum)
{
  int i = blockIdx.x * blockDim.x + threadIdx.x;
  if (i >= N_EDGES * NHEAD) return;
  int e = i >> 2, h = i & 3;
  int d = ei[N_EDGES + e];
  float m = funkey(mmax[d * NHEAD + h]);
  float a = __expf(sc[i] - m);
  aex[i] = a;
  atomicAdd(&zsum[d * NHEAD + h], a);     // segment sum
}

__global__ void edge_agg_kernel(const int* __restrict__ ei,
                                const float* __restrict__ aex,
                                const float* __restrict__ zsum,
                                const float* __restrict__ v,
                                float* __restrict__ agg)
{
  int i = blockIdx.x * blockDim.x + threadIdx.x;   // E * NHEAD * 64 chunks of 4
  if (i >= N_EDGES * NHEAD * 64) return;
  int e   = i >> 8;
  int rem = i & 255;
  int h   = rem >> 6;
  int d0  = (rem & 63) * 4;
  int s = ei[e], dd = ei[N_EDGES + e];
  float w = aex[e * NHEAD + h] / (zsum[dd * NHEAD + h] + 1e-16f);
  const float* vp = v + (long)s * HD + h * DHEADC + d0;
  float* ap = agg + (long)dd * HD + h * DHEADC + d0;
  #pragma unroll
  for (int u = 0; u < 4; ++u) atomicAdd(ap + u, w * vp[u]);
}

__global__ void tconv_finish_kernel(const float* __restrict__ tr,
                                    const float* __restrict__ agg,
                                    const float* __restrict__ pe_row,
                                    float* __restrict__ out)
{
  int i = blockIdx.x * blockDim.x + threadIdx.x;
  if (i >= N_NODES * DMODEL) return;
  int n = i >> 8, c = i & 255;
  const float* ap = agg + (long)n * HD + c;
  float m = 0.25f * (ap[0] + ap[256] + ap[512] + ap[768]);  // mean over heads
  float v = tr[i] + m;
  if (pe_row) v += pe_row[c];
  out[i] = v;
}

// ---------------- temporal attention, last query row only ----------------
__global__ __launch_bounds__(256) void attn_last_kernel(
    const float* __restrict__ qlast, const float* __restrict__ kc,
    const float* __restrict__ vc, float* __restrict__ oattn, int T)
{
  int g    = blockIdx.x * 8 + (threadIdx.x >> 5);   // (node, head)
  int lane = threadIdx.x & 31;
  int b = g >> 2, h = g & 3;
  if (b >= N_NODES) return;
  const float* qp = qlast + (long)b * DMODEL + h * DH;
  float s = -3.0e38f;
  if (lane < T) {
    const float* kp = kc + ((long)lane * N_NODES + b) * DMODEL + h * DH;
    float p = 0.0f;
    #pragma unroll
    for (int d = 0; d < DH; ++d) p += qp[d] * kp[d];
    s = p * 0.125f;                       // 1/sqrt(64)
  }
  float mx = s;
  #pragma unroll
  for (int m = 16; m > 0; m >>= 1) mx = fmaxf(mx, __shfl_xor(mx, m, 32));
  float p = __expf(s - mx);
  float zs = p;
  #pragma unroll
  for (int m = 16; m > 0; m >>= 1) zs += __shfl_xor(zs, m, 32);
  float att = p / zs;
  int d0 = lane * 2;
  float o0 = 0.0f, o1 = 0.0f;
  for (int t = 0; t < T; ++t) {
    float w = __shfl(att, t, 32);
    const float* vp = vc + ((long)t * N_NODES + b) * DMODEL + h * DH + d0;
    o0 += w * vp[0];
    o1 += w * vp[1];
  }
  float* op = oattn + (long)b * DMODEL + h * DH + d0;
  op[0] = o0; op[1] = o1;
}

// ---------------- layernorm over 256, one wave per row ----------------
__global__ __launch_bounds__(256) void layernorm_kernel(
    const float* __restrict__ x, const float* __restrict__ g,
    const float* __restrict__ b, float* __restrict__ out, int Mrows)
{
  int row  = blockIdx.x * 8 + (threadIdx.x >> 5);
  int lane = threadIdx.x & 31;
  if (row >= Mrows) return;
  const float* xp = x + (long)row * DMODEL;
  float vals[8]; float s = 0.0f;
  #pragma unroll
  for (int j = 0; j < 8; ++j) { vals[j] = xp[lane + 32 * j]; s += vals[j]; }
  #pragma unroll
  for (int m = 16; m > 0; m >>= 1) s += __shfl_xor(s, m, 32);
  float mu = s * (1.0f / 256.0f);
  float vs = 0.0f;
  #pragma unroll
  for (int j = 0; j < 8; ++j) { float d = vals[j] - mu; vs += d * d; }
  #pragma unroll
  for (int m = 16; m > 0; m >>= 1) vs += __shfl_xor(vs, m, 32);
  float rs = rsqrtf(vs * (1.0f / 256.0f) + 1e-5f);
  float* op = out + (long)row * DMODEL;
  #pragma unroll
  for (int j = 0; j < 8; ++j) { int c = lane + 32 * j; op[c] = (vals[j] - mu) * rs * g[c] + b[c]; }
}

__global__ void latlon_kernel(const float* __restrict__ known_x, float* __restrict__ xnext)
{
  int n = blockIdx.x * blockDim.x + threadIdx.x;
  if (n >= N_NODES) return;
  const float* src = known_x + ((long)(KKNOWN - 1) * N_NODES + n) * FEAT;
  xnext[n * FEAT + 0] = src[0];
  xnext[n * FEAT + 1] = src[1];
}

// ---------------- host orchestration ----------------
extern "C" void kernel_launch(void* const* d_in, const int* in_sizes, int n_in,
                              void* d_out, int out_size, void* d_ws, size_t ws_size,
                              hipStream_t stream)
{
  const float* known_x = (const float*)d_in[0];
  const int*   ei      = (const int*)d_in[1];
  const float* Wq0 = (const float*)d_in[2];  const float* bq0 = (const float*)d_in[3];
  const float* Wk0 = (const float*)d_in[4];  const float* bk0 = (const float*)d_in[5];
  const float* Wv0 = (const float*)d_in[6];  const float* bv0 = (const float*)d_in[7];
  const float* Wr0 = (const float*)d_in[8];  const float* br0 = (const float*)d_in[9];
  const float* Wq1 = (const float*)d_in[10]; const float* bq1 = (const float*)d_in[11];
  const float* Wk1 = (const float*)d_in[12]; const float* bk1 = (const float*)d_in[13];
  const float* Wv1 = (const float*)d_in[14]; const float* bv1 = (const float*)d_in[15];
  const float* Wr1 = (const float*)d_in[16]; const float* br1 = (const float*)d_in[17];
  const float* pe   = (const float*)d_in[18];
  const float* Wqkv = (const float*)d_in[19]; const float* bqkv = (const float*)d_in[20];
  const float* Wo   = (const float*)d_in[21]; const float* bo   = (const float*)d_in[22];
  const float* W1   = (const float*)d_in[23]; const float* b1f  = (const float*)d_in[24];
  const float* W2   = (const float*)d_in[25]; const float* b2f  = (const float*)d_in[26];
  const float* Wd   = (const float*)d_in[27]; const float* bd   = (const float*)d_in[28];
  const float* g1   = (const float*)d_in[29]; const float* be1  = (const float*)d_in[30];
  const float* g2   = (const float*)d_in[31]; const float* be2  = (const float*)d_in[32];

  // workspace carve-out (floats; all chunk sizes multiples of 4 -> 16B aligned)
  float* ws = (float*)d_ws;
  size_t off = 0;
  auto alloc = [&](size_t n) { float* p = ws + off; off += n; return p; };
  const size_t NM = (size_t)N_NODES * DMODEL;
  float* hid   = alloc((size_t)NGEN * NM);   // hidden[t]
  float* kc    = alloc((size_t)NGEN * NM);   // K cache
  float* vc    = alloc((size_t)NGEN * NM);   // V cache
  float* tq    = alloc((size_t)N_NODES * HD);
  float* tk    = alloc((size_t)N_NODES * HD);
  float* tv    = alloc((size_t)N_NODES * HD);
  float* agg   = alloc((size_t)N_NODES * HD);
  float* tr    = alloc(NM);
  float* sc    = alloc((size_t)N_EDGES * NHEAD);
  float* aex   = alloc((size_t)N_EDGES * NHEAD);
  float* zsum  = alloc((size_t)N_NODES * NHEAD);
  unsigned* mmax = (unsigned*)alloc((size_t)N_NODES * NHEAD);
  float* h1    = alloc(NM);
  float* qlast = alloc(NM);
  float* oattn = alloc(NM);
  float* h2tmp = alloc(NM);
  float* hln1  = alloc(NM);
  float* ff1   = alloc((size_t)N_NODES * FFDIM);
  float* ctx   = alloc(NM);
  float* xnext = alloc((size_t)(N_NODES * FEAT + 8));
  // bf16 transposed weights (ushort counts are even -> allocate half as floats)
  auto allocb = [&](size_t elems) { return (unsigned short*)alloc(elems / 2); };
  unsigned short* btWq1 = allocb((size_t)HD * DMODEL);
  unsigned short* btWk1 = allocb((size_t)HD * DMODEL);
  unsigned short* btWv1 = allocb((size_t)HD * DMODEL);
  unsigned short* btWr1 = allocb((size_t)DMODEL * DMODEL);
  unsigned short* btQ   = allocb((size_t)DMODEL * DMODEL);
  unsigned short* btK   = allocb((size_t)DMODEL * DMODEL);
  unsigned short* btV   = allocb((size_t)DMODEL * DMODEL);
  unsigned short* btWo  = allocb((size_t)DMODEL * DMODEL);
  unsigned short* btW1  = allocb((size_t)FFDIM * DMODEL);
  unsigned short* btW2  = allocb((size_t)DMODEL * FFDIM);

  auto wconv = [&](const float* W, int ldw, int rows, int cols,
                   unsigned short* Bt, int Kpad, int Npad) {
    int tot = Npad * Kpad;
    wconvert_kernel<<<(tot + 255) / 256, 256, 0, stream>>>(W, ldw, rows, cols, Bt, Kpad, Npad);
  };
  // one-time (per launch) weight conversion
  wconv(Wq1, HD, DMODEL, HD, btWq1, DMODEL, HD);
  wconv(Wk1, HD, DMODEL, HD, btWk1, DMODEL, HD);
  wconv(Wv1, HD, DMODEL, HD, btWv1, DMODEL, HD);
  wconv(Wr1, DMODEL, DMODEL, DMODEL, btWr1, DMODEL, DMODEL);
  wconv(Wqkv,              3 * DMODEL, DMODEL, DMODEL, btQ, DMODEL, DMODEL);
  wconv(Wqkv + DMODEL,     3 * DMODEL, DMODEL, DMODEL, btK, DMODEL, DMODEL);
  wconv(Wqkv + 2 * DMODEL, 3 * DMODEL, DMODEL, DMODEL, btV, DMODEL, DMODEL);
  wconv(Wo, DMODEL, DMODEL, DMODEL, btWo, DMODEL, DMODEL);
  wconv(W1, FFDIM, DMODEL, FFDIM, btW1, DMODEL, FFDIM);
  wconv(W2, DMODEL, FFDIM, DMODEL, btW2, FFDIM, DMODEL);

  // fast GEMM dispatch: template specializations for the 3 epilogue variants used
  auto gemmF = [&](const float* A, int lda, const unsigned short* Bt, int Kpad,
                   const float* bias, float* C, int ldc, int M, int Nc, int Kd) {
    dim3 grid(Nc / 128, M / 64);
    gemm_bf16_fast<false, false><<<grid, dim3(256), 0, stream>>>(
        A, lda, Bt, Kpad, bias, nullptr, 0, C, ldc, Kd);
  };
  auto gemmF_add = [&](const float* A, int lda, const unsigned short* Bt, int Kpad,
                       const float* bias, const float* add, int ldadd,
                       float* C, int ldc, int M, int Nc, int Kd) {
    dim3 grid(Nc / 128, M / 64);
    gemm_bf16_fast<true, false><<<grid, dim3(256), 0, stream>>>(
        A, lda, Bt, Kpad, bias, add, ldadd, C, ldc, Kd);
  };
  auto gemmF_relu = [&](const float* A, int lda, const unsigned short* Bt, int Kpad,
                        const float* bias, float* C, int ldc, int M, int Nc, int Kd) {
    dim3 grid(Nc / 128, M / 64);
    gemm_bf16_fast<false, true><<<grid, dim3(256), 0, stream>>>(
        A, lda, Bt, Kpad, bias, nullptr, 0, C, ldc, Kd);
  };
  auto gemmG = [&](const float* A, int lda, const float* B, int ldb,
                   const float* bias, float* C, int ldc, int M, int Nc, int Kd) {
    dim3 grid((Nc + 127) / 128, (M + 63) / 64);
    gemm_bf16_generic<<<grid, dim3(256), 0, stream>>>(A, lda, B, ldb, bias, C, ldc, M, Nc, Kd);
  };

  auto edge_phase = [&](const float* qv, const float* kv, const float* vv) {
    hipMemsetAsync(mmax, 0, (size_t)N_NODES * NHEAD * sizeof(unsigned), stream);
    hipMemsetAsync(zsum, 0, (size_t)N_NODES * NHEAD * sizeof(float), stream);
    hipMemsetAsync(agg,  0, (size_t)N_NODES * HD * sizeof(float), stream);
    edge_dot_kernel<<<N_EDGES / 8, 256, 0, stream>>>(ei, qv, kv, sc, mmax);
    edge_exp_kernel<<<(N_EDGES * NHEAD + 255) / 256, 256, 0, stream>>>(ei, sc, mmax, aex, zsum);
    edge_agg_kernel<<<(N_EDGES * NHEAD * 64 + 255) / 256, 256, 0, stream>>>(ei, aex, zsum, vv, agg);
  };

  auto spatial = [&](const float* x, int lda, int tstep) {
    // layer 0 (fin = 10): ragged K -> generic GEMMs (tiny fraction of FLOPs)
    gemmG(x, lda, Wq0, HD,     bq0, tq, HD,     N_NODES, HD,     FEAT);
    gemmG(x, lda, Wk0, HD,     bk0, tk, HD,     N_NODES, HD,     FEAT);
    gemmG(x, lda, Wv0, HD,     bv0, tv, HD,     N_NODES, HD,     FEAT);
    gemmG(x, lda, Wr0, DMODEL, br0, tr, DMODEL, N_NODES, DMODEL, FEAT);
    edge_phase(tq, tk, tv);
    tconv_finish_kernel<<<(N_NODES * DMODEL + 255) / 256, 256, 0, stream>>>(tr, agg, nullptr, h1);
    // layer 1 (fin = 256): fast WMMA GEMMs on pre-converted weights
    float* ho = hid + (size_t)tstep * NM;
    gemmF(h1, DMODEL, btWq1, DMODEL, bq1, tq, HD,     N_NODES, HD,     DMODEL);
    gemmF(h1, DMODEL, btWk1, DMODEL, bk1, tk, HD,     N_NODES, HD,     DMODEL);
    gemmF(h1, DMODEL, btWv1, DMODEL, bv1, tv, HD,     N_NODES, HD,     DMODEL);
    gemmF(h1, DMODEL, btWr1, DMODEL, br1, tr, DMODEL, N_NODES, DMODEL, DMODEL);
    edge_phase(tq, tk, tv);
    tconv_finish_kernel<<<(N_NODES * DMODEL + 255) / 256, 256, 0, stream>>>(
        tr, agg, pe + (size_t)tstep * DMODEL, ho);
    // KV cache projections for this timestep
    gemmF(ho, DMODEL, btK, DMODEL, bqkv + DMODEL,
          kc + (size_t)tstep * NM, DMODEL, N_NODES, DMODEL, DMODEL);
    gemmF(ho, DMODEL, btV, DMODEL, bqkv + 2 * DMODEL,
          vc + (size_t)tstep * NM, DMODEL, N_NODES, DMODEL, DMODEL);
  };

  // ---- known timesteps ----
  for (int i = 0; i < KKNOWN; ++i)
    spatial(known_x + (size_t)i * N_NODES * FEAT, FEAT, i);

  // ---- autoregressive generation ----
  for (int t = KKNOWN; t < NGEN; ++t) {
    const int T = t;
    const float* hlast = hid + (size_t)(T - 1) * NM;
    gemmF(hlast, DMODEL, btQ, DMODEL, bqkv, qlast, DMODEL, N_NODES, DMODEL, DMODEL);
    attn_last_kernel<<<(N_NODES * NHEAD) / 8, 256, 0, stream>>>(qlast, kc, vc, oattn, T);
    gemmF_add(oattn, DMODEL, btWo, DMODEL, bo, hlast, DMODEL, h2tmp, DMODEL,
              N_NODES, DMODEL, DMODEL);
    layernorm_kernel<<<N_NODES / 8, 256, 0, stream>>>(h2tmp, g1, be1, hln1, N_NODES);
    gemmF_relu(hln1, DMODEL, btW1, DMODEL, b1f, ff1, FFDIM, N_NODES, FFDIM, DMODEL);
    gemmF_add(ff1, FFDIM, btW2, FFDIM, b2f, hln1, DMODEL, h2tmp, DMODEL,
              N_NODES, DMODEL, FFDIM);
    layernorm_kernel<<<N_NODES / 8, 256, 0, stream>>>(h2tmp, g2, be2, ctx, N_NODES);
    // x_next = [latlon, ctx @ Wd + bd]  (N=8 -> generic)
    gemmG(ctx, DMODEL, Wd, FEAT - 2, bd, xnext + 2, FEAT, N_NODES, FEAT - 2, DMODEL);
    latlon_kernel<<<(N_NODES + 255) / 256, 256, 0, stream>>>(known_x, xnext);
    hipMemcpyAsync((float*)d_out + (size_t)(t - KKNOWN) * N_NODES * FEAT, xnext,
                   (size_t)N_NODES * FEAT * sizeof(float),
                   hipMemcpyDeviceToDevice, stream);
    spatial(xnext, FEAT, t);
  }
}
